// VectorQuantizer_17592186045165
// MI455X (gfx1250) — matrix operand order repaired and checked by reading, not compiled, not run on hardware
//
#include <hip/hip_runtime.h>

typedef _Float16 f16;
typedef _Float16 v16h __attribute__((ext_vector_type(16)));
typedef _Float16 v8h  __attribute__((ext_vector_type(8)));
typedef float    v8f  __attribute__((ext_vector_type(8)));

#define V_   16
#define N_   8192
#define D_   256
#define K_   1024
#define ROWS 64          // N rows per block
#define KC   64          // K columns per chunk
#define AS   136         // padded LDS stride in halves (272B = 68 dwords -> 4-dword bank rotation)

union V16U { v16h v; v8h h[2]; };

__device__ __forceinline__ v8f wmma_f16(v16h a, v16h b, v8f c) {
  // v_wmma_f32_16x16x32_f16: (neg_a, A, neg_b, B, c_mod, C, reuse_a, reuse_b)
  return __builtin_amdgcn_wmma_f32_16x16x32_f16(false, a, false, b, (short)0, c, false, false);
}

// ---------------- Kernel 1: per-column squared norms of W ----------------
__global__ __launch_bounds__(256)
void vq_wsq_kernel(const float* __restrict__ W, float* __restrict__ wsq) {
  const int v = blockIdx.y;
  const int k = blockIdx.x * 256 + threadIdx.x;
  const float* Wp = W + (size_t)v * D_ * K_ + k;
  float s = 0.f;
  for (int d = 0; d < D_; ++d) {
    float x = Wp[(size_t)d * K_];
    s += x * x;
  }
  wsq[v * K_ + k] = s;
}

// ---------------- Kernel 2: fused split-f16 WMMA GEMM + argmin + gather ----------------
__global__ __launch_bounds__(256)
void vq_main_kernel(const float* __restrict__ X, const float* __restrict__ W,
                    const float* __restrict__ wsq, float* __restrict__ Out,
                    float* __restrict__ partial) {
  __shared__ __align__(16) char smem[35840];
  __shared__ int bestk[ROWS];

  f16*   Ahi  = (f16*)smem;              // 64 x AS halves (17408 B) ; aliased as B-hi
  f16*   Alo  = (f16*)(smem + 17408);    // 64 x AS halves          ; aliased as B-lo
  float* fmin = (float*)smem;            // [64][64] (16384 B), after compute
  int*   imin = (int*)(smem + 16384);    // [64][64] (16384 B)
  float* lred = (float*)smem;            // [256] loss reduction

  const int tid  = threadIdx.x;
  const int lane = tid & 31;
  const int wave = tid >> 5;
  const int wr   = wave & 3;             // row-tile (0..3) -> rows wr*16..+15
  const int wc   = wave >> 2;            // column-tile pair (0..1)
  const int l15  = lane & 15;
  const int hi16 = lane >> 4;            // 0 for lanes 0-15, 1 for lanes 16-31

  const int v  = blockIdx.y;
  const int n0 = blockIdx.x * ROWS;

  const float* Xg = X + ((size_t)v * N_ + n0) * D_;
  const float* Wg = W + (size_t)v * D_ * K_;
  const float* Qg = wsq + (size_t)v * K_;

  v16h ah[8], al[8];

  // ---- stage A tile (fp32 -> f16 hi/lo split) in two D-halves, build fragments ----
  #pragma unroll
  for (int h = 0; h < 2; ++h) {
    __syncthreads();
    #pragma unroll
    for (int i = 0; i < 8; ++i) {
      int f   = i * 256 + tid;           // 2048 float4 per half
      int row = f >> 5;
      int q   = f & 31;
      const float4 x4 = *(const float4*)(Xg + (size_t)row * D_ + h * 128 + q * 4);
      float xs[4] = {x4.x, x4.y, x4.z, x4.w};
      #pragma unroll
      for (int j = 0; j < 4; ++j) {
        f16 hp = (f16)xs[j];
        f16 lp = (f16)(xs[j] - (float)hp);
        Ahi[row * AS + q * 4 + j] = hp;
        Alo[row * AS + q * 4 + j] = lp;
      }
    }
    __syncthreads();
    // A fragment (16-bit A 16x32): lanes hold K kb..kb+15 contiguous, kb = hi16*16
    #pragma unroll
    for (int t = 0; t < 4; ++t) {
      int off = (wr * 16 + l15) * AS + t * 32 + hi16 * 16;
      V16U uh, ul;
      uh.h[0] = *(const v8h*)(Ahi + off);
      uh.h[1] = *(const v8h*)(Ahi + off + 8);
      ul.h[0] = *(const v8h*)(Alo + off);
      ul.h[1] = *(const v8h*)(Alo + off + 8);
      ah[h * 4 + t] = uh.v;
      al[h * 4 + t] = ul.v;
    }
  }

  float mval[2][8];
  int   midx[2][8];
  #pragma unroll
  for (int c = 0; c < 2; ++c)
    #pragma unroll
    for (int j = 0; j < 8; ++j) { mval[c][j] = 3.4e38f; midx[c][j] = 0; }

  const v8f vzero = {0.f, 0.f, 0.f, 0.f, 0.f, 0.f, 0.f, 0.f};

  // ---- loop over K in chunks of KC columns ----
  for (int kc = 0; kc < K_ / KC; ++kc) {
    v8f acc0 = vzero;
    v8f acc1 = vzero;

    #pragma unroll
    for (int h = 0; h < 2; ++h) {
      __syncthreads();
      // stage W chunk transposed: Bt[kcol][d'] for d' = 0..127 of this half
      #pragma unroll
      for (int i = 0; i < 8; ++i) {
        int f  = i * 256 + tid;          // 2048 float4 per half (128 d x 16 f4 along k)
        int d  = f >> 4;
        int kq = f & 15;
        const float4 w4 = *(const float4*)(Wg + (size_t)(h * 128 + d) * K_ + kc * KC + kq * 4);
        float ws4[4] = {w4.x, w4.y, w4.z, w4.w};
        #pragma unroll
        for (int j = 0; j < 4; ++j) {
          int kcol = kq * 4 + j;
          f16 hp = (f16)ws4[j];
          f16 lp = (f16)(ws4[j] - (float)hp);
          Ahi[kcol * AS + d] = hp;       // aliased B-hi
          Alo[kcol * AS + d] = lp;       // aliased B-lo
        }
      }
      __syncthreads();
      // interleave the two independent column-tile accumulator chains so that
      // adjacent v_wmma instructions never share a D->C RAW dependency
      #pragma unroll
      for (int t = 0; t < 4; ++t) {
        int off0 = ((wc * 2 + 0) * 16 + l15) * AS + t * 32 + hi16 * 16;
        int off1 = ((wc * 2 + 1) * 16 + l15) * AS + t * 32 + hi16 * 16;
        V16U bh0, bl0, bh1, bl1;
        bh0.h[0] = *(const v8h*)(Ahi + off0);
        bh0.h[1] = *(const v8h*)(Ahi + off0 + 8);
        bh1.h[0] = *(const v8h*)(Ahi + off1);
        bh1.h[1] = *(const v8h*)(Ahi + off1 + 8);
        bl0.h[0] = *(const v8h*)(Alo + off0);
        bl0.h[1] = *(const v8h*)(Alo + off0 + 8);
        bl1.h[0] = *(const v8h*)(Alo + off1);
        bl1.h[1] = *(const v8h*)(Alo + off1 + 8);
        int ht = h * 4 + t;
        acc0 = wmma_f16(ah[ht], bh0.v, acc0);  // hi*hi (tile 0)
        acc1 = wmma_f16(ah[ht], bh1.v, acc1);  // hi*hi (tile 1)
        acc0 = wmma_f16(ah[ht], bl0.v, acc0);  // hi*lo (tile 0)
        acc1 = wmma_f16(ah[ht], bl1.v, acc1);  // hi*lo (tile 1)
        acc0 = wmma_f16(al[ht], bh0.v, acc0);  // lo*hi (tile 0)
        acc1 = wmma_f16(al[ht], bh1.v, acc1);  // lo*hi (tile 1)
      }
    }
    // fused argmin update: dist(k) = ||w_k||^2 - 2*s  (x^2 term is row-constant)
    #pragma unroll
    for (int c = 0; c < 2; ++c) {
      int kg = kc * KC + (wc * 2 + c) * 16 + l15;
      float wq = Qg[kg];
      #pragma unroll
      for (int j = 0; j < 8; ++j) {
        float s    = (c == 0) ? acc0[j] : acc1[j];
        float dist = wq - 2.0f * s;
        if (dist < mval[c][j]) { mval[c][j] = dist; midx[c][j] = kg; }
      }
    }
  }

  // ---- cross-lane / cross-wave argmin via LDS (slot order == ascending k) ----
  __syncthreads();
  #pragma unroll
  for (int c = 0; c < 2; ++c) {
    int slot = (wc * 2 + c) * 16 + l15;
    #pragma unroll
    for (int j = 0; j < 8; ++j) {
      int row = wr * 16 + hi16 * 8 + j;  // C layout: lanes 16-31 hold M = 8+j
      fmin[row * 64 + slot] = mval[c][j];
      imin[row * 64 + slot] = midx[c][j];
    }
  }
  __syncthreads();
  if (tid < ROWS) {
    float b = 3.4e38f; int bi = 0;
    for (int s = 0; s < 64; ++s) {
      float x = fmin[tid * 64 + s];
      if (x < b) { b = x; bi = imin[tid * 64 + s]; }
    }
    bestk[tid] = bi;
  }
  __syncthreads();

  // ---- epilogue: gather winning codeword, write straight-through output, loss ----
  float lsum = 0.f;
  for (int rr = 0; rr < 8; ++rr) {
    int row  = wave + rr * 8;
    int kidx = bestk[row];
    const float* xr   = Xg + (size_t)row * D_;
    float*       outr = Out + ((size_t)v * N_ + n0 + row) * D_;
    #pragma unroll
    for (int e = 0; e < 8; ++e) {
      int d = e * 32 + lane;
      float q   = Wg[(size_t)d * K_ + kidx];
      float x   = xr[d];
      float dif = q - x;
      outr[d] = x + dif;                 // match reference's x + (q - x) rounding
      lsum += dif * dif;
    }
  }
  lred[tid] = lsum;
  __syncthreads();
  for (int s = 128; s > 0; s >>= 1) {
    if (tid < s) lred[tid] += lred[tid + s];
    __syncthreads();
  }
  if (tid == 0) partial[blockIdx.y * gridDim.x + blockIdx.x] = lred[0];
}

// ---------------- Kernel 3: final loss reduction ----------------
__global__ __launch_bounds__(256)
void vq_loss_kernel(const float* __restrict__ partial, float* __restrict__ out_loss) {
  __shared__ float red[256];
  float s = 0.f;
  for (int i = threadIdx.x; i < (N_ / ROWS) * V_; i += 256) s += partial[i];
  red[threadIdx.x] = s;
  __syncthreads();
  for (int t = 128; t > 0; t >>= 1) {
    if (threadIdx.x < t) red[threadIdx.x] += red[threadIdx.x + t];
    __syncthreads();
  }
  if (threadIdx.x == 0)
    out_loss[0] = red[0] * 1.25f / (float)((size_t)V_ * N_ * D_);  // q_loss + 0.25*e_loss
}

extern "C" void kernel_launch(void* const* d_in, const int* in_sizes, int n_in,
                              void* d_out, int out_size, void* d_ws, size_t ws_size,
                              hipStream_t stream) {
  const float* X = (const float*)d_in[0];   // [V, N, D] fp32
  const float* W = (const float*)d_in[1];   // [V, D, K] fp32
  float* Out     = (float*)d_out;           // [V, N, D] then loss scalar
  float* wsq     = (float*)d_ws;            // V*K floats
  float* partial = wsq + V_ * K_;           // (N/ROWS)*V floats
  float* loss    = Out + (size_t)V_ * N_ * D_;

  vq_wsq_kernel<<<dim3(K_ / 256, V_), 256, 0, stream>>>(W, wsq);
  vq_main_kernel<<<dim3(N_ / ROWS, V_), 256, 0, stream>>>(X, W, wsq, Out, partial);
  vq_loss_kernel<<<1, 256, 0, stream>>>(partial, loss);
}
